// Multihead_5849745457483
// MI455X (gfx1250) — compile-verified, matrix operand order checked by reading, not run on hardware
//
#include <hip/hip_runtime.h>

typedef __attribute__((ext_vector_type(16))) _Float16 v16h;
typedef __attribute__((ext_vector_type(4)))  _Float16 v4h;
typedef __attribute__((ext_vector_type(8)))  float    v8f;

#define WMMA_F16(a, b, c) \
  __builtin_amdgcn_wmma_f32_16x16x32_f16(false, (a), false, (b), (short)0, (c), false, false)

// Problem constants
#define DMODEL 1024
#define SEQLEN 2048
#define NBATCH 2
#define NHEADS 16
#define DK     64

// ---------------------------------------------------------------------------
// Fragment loader (wave32).
// A-matrix 16x32 f16 layout: lane L -> row m = L&15, half = L>>4;
//   halfs 0..7  : K = half*8 + i
//   halfs 8..15 : K = 16 + half*8 + i
// The same loader produces a B-fragment when the B matrix (K x 16) is stored
// as [n][k] row-major (i.e. column n of B is a contiguous K-run).
// ---------------------------------------------------------------------------
__device__ __forceinline__ v16h ld_frag(const _Float16* base, int row0, int ld, int kbase) {
  const int lane = threadIdx.x & 31;
  const int m  = lane & 15;
  const int lh = lane >> 4;
  const _Float16* p = base + (size_t)(row0 + m) * ld + kbase + lh * 8;
  v4h a0 = *(const v4h*)(p + 0);
  v4h a1 = *(const v4h*)(p + 4);
  v4h b0 = *(const v4h*)(p + 16);
  v4h b1 = *(const v4h*)(p + 20);
  v16h r;
#pragma unroll
  for (int i = 0; i < 4; ++i) {
    r[i] = a0[i]; r[4 + i] = a1[i]; r[8 + i] = b0[i]; r[12 + i] = b1[i];
  }
  return r;
}

// 16 fp32 -> 16 f16 into LDS (global->LDS staging with conversion)
__device__ __forceinline__ void cvt_store16(_Float16* dst, const float4* src) {
  float4 f0 = src[0], f1 = src[1], f2 = src[2], f3 = src[3];
  dst[0]  = (_Float16)f0.x; dst[1]  = (_Float16)f0.y; dst[2]  = (_Float16)f0.z; dst[3]  = (_Float16)f0.w;
  dst[4]  = (_Float16)f1.x; dst[5]  = (_Float16)f1.y; dst[6]  = (_Float16)f1.z; dst[7]  = (_Float16)f1.w;
  dst[8]  = (_Float16)f2.x; dst[9]  = (_Float16)f2.y; dst[10] = (_Float16)f2.z; dst[11] = (_Float16)f2.w;
  dst[12] = (_Float16)f3.x; dst[13] = (_Float16)f3.y; dst[14] = (_Float16)f3.z; dst[15] = (_Float16)f3.w;
}

// ---------------------------------------------------------------------------
// CDNA5 async global->LDS copy of 16 halfs (two B128 transfers, ASYNCcnt).
// LDS destination offset = low 32 bits of the generic pointer (LDS aperture).
// Destination must be 16B aligned.
// ---------------------------------------------------------------------------
__device__ __forceinline__ void async_cp16h(_Float16* ldst, const _Float16* gsrc) {
  const uint32_t l = (uint32_t)(uintptr_t)ldst;
  const uint64_t g = (uint64_t)(uintptr_t)gsrc;
  asm volatile("global_load_async_to_lds_b128 %0, %1, off"
               :: "v"(l), "v"(g) : "memory");
  asm volatile("global_load_async_to_lds_b128 %0, %1, off"
               :: "v"(l + 16u), "v"(g + 16u) : "memory");
}

__device__ __forceinline__ void wait_async0() {
  asm volatile("s_wait_asynccnt 0" ::: "memory");
}

// ---------------------------------------------------------------------------
// Kernel 1: qkv = x @ Wqkv^T + bqkv   (M=4096, N=3072, K=1024)
// Both operands are fp32 -> converting staging path (VALU cvt into LDS).
// Scatters result into head-major f16 buffers:
//   Qh,Kh : [B,H,S,64]   Vth : [B,H,64,S]
// ---------------------------------------------------------------------------
__global__ __launch_bounds__(128) void qkv_gemm_kernel(
    const float* __restrict__ x, const float* __restrict__ Wqkv,
    const float* __restrict__ bqkv,
    _Float16* __restrict__ Qh, _Float16* __restrict__ Kh, _Float16* __restrict__ Vth) {
  __shared__ _Float16 At[64 * 40];
  __shared__ _Float16 Bt[64 * 40];
  const int tid = threadIdx.x, wave = tid >> 5, lane = tid & 31;
  const int m0 = blockIdx.x * 64, n0 = blockIdx.y * 64;
  const int lrow = tid >> 1, lseg = tid & 1;
  v8f acc[4] = {};

  for (int k0 = 0; k0 < DMODEL; k0 += 32) {
    __syncthreads();
    cvt_store16(&At[lrow * 40 + lseg * 16],
                (const float4*)(x + (size_t)(m0 + lrow) * DMODEL + k0 + lseg * 16));
    cvt_store16(&Bt[lrow * 40 + lseg * 16],
                (const float4*)(Wqkv + (size_t)(n0 + lrow) * DMODEL + k0 + lseg * 16));
    __syncthreads();
    v16h a = ld_frag(At, wave * 16, 40, 0);
#pragma unroll
    for (int j = 0; j < 4; ++j) {
      v16h b = ld_frag(Bt, j * 16, 40, 0);
      acc[j] = WMMA_F16(a, b, acc[j]);
    }
  }

  const int lh = lane >> 4, nl = lane & 15;
#pragma unroll
  for (int j = 0; j < 4; ++j) {
    const int n = n0 + j * 16 + nl;
    const float bias = bqkv[n];
    const int h = n / 192, rr = n % 192;
#pragma unroll
    for (int r = 0; r < 8; ++r) {
      const int m = m0 + wave * 16 + r + 8 * lh;
      const int bb = m >> 11, s = m & (SEQLEN - 1);
      const _Float16 hv = (_Float16)(acc[j][r] + bias);
      const size_t bh = (size_t)(bb * NHEADS + h);
      if (rr < 64)        Qh[(bh * SEQLEN + s) * DK + rr] = hv;
      else if (rr < 128)  Kh[(bh * SEQLEN + s) * DK + (rr - 64)] = hv;
      else                Vth[(bh * DK + (rr - 128)) * SEQLEN + s] = hv;
    }
  }
}

// ---------------------------------------------------------------------------
// Kernel 2: flash attention.  grid = B*H*(S/64), 4 waves, wave = 16 q-rows.
// Double-buffered K/V tiles filled with async global->LDS copies (ASYNCcnt):
// next kv tile streams into LDS while WMMAs consume the current one.
// ---------------------------------------------------------------------------
__global__ __launch_bounds__(128) void attn_kernel(
    const _Float16* __restrict__ Qh, const _Float16* __restrict__ Kh,
    const _Float16* __restrict__ Vth, _Float16* __restrict__ Oh) {
  __shared__ _Float16 Kt[2][32 * 72];   // [kv][dk], pitch 72 halfs (144B)
  __shared__ _Float16 Vt[2][64 * 48];   // [dk][kv], pitch 48 halfs (96B, 16B aligned)
  __shared__ _Float16 Pw[4][16 * 40];   // per-wave P transpose scratch

  const int tid = threadIdx.x, wave = tid >> 5, lane = tid & 31;
  const int lh = lane >> 4, nl = lane & 15;
  const int qt = blockIdx.x & 31;
  const int h  = (blockIdx.x >> 5) & 15;
  const int b  = blockIdx.x >> 9;
  const size_t bh = (size_t)(b * NHEADS + h);

  const _Float16* Qbase = Qh + (bh * SEQLEN + qt * 64 + wave * 16) * DK;
  const _Float16* Kgb   = Kh + bh * SEQLEN * DK;
  const _Float16* Vgb   = Vth + bh * DK * SEQLEN;

  const v16h qa0 = ld_frag(Qbase, 0, DK, 0);
  const v16h qa1 = ld_frag(Qbase, 0, DK, 32);

  v8f o0 = {}, o1 = {}, o2 = {}, o3 = {};
  float mrow[8], lsum[8];
#pragma unroll
  for (int r = 0; r < 8; ++r) { mrow[r] = -1e30f; lsum[r] = 0.0f; }

  const int krow = tid >> 2, kq = tid & 3;   // K tile: 32 rows x 64, 16h per thread
  const int vrow = tid >> 1, vs = tid & 1;   // V tile: 64 rows x 32, 16h per thread

  auto issue_tile = [&](int kt, int buf) {
    const int kv0 = kt * 32;
    async_cp16h(&Kt[buf][krow * 72 + kq * 16], Kgb + (size_t)(kv0 + krow) * DK + kq * 16);
    async_cp16h(&Vt[buf][vrow * 48 + vs * 16], Vgb + (size_t)vrow * SEQLEN + kv0 + vs * 16);
  };

  constexpr int NT = SEQLEN / 32;
  issue_tile(0, 0);
  wait_async0();
  __syncthreads();

  for (int kt = 0; kt < NT; ++kt) {
    const int buf = kt & 1;
    if (kt + 1 < NT) issue_tile(kt + 1, buf ^ 1);   // prefetch next tile (async)

    // S = Q * K^T  (16 x 32) : two 16x16 C tiles, each 2 K-steps
    v8f s0 = {}, s1 = {};
    {
      v16h b0 = ld_frag(Kt[buf], 0, 72, 0);   s0 = WMMA_F16(qa0, b0, s0);
      v16h b1 = ld_frag(Kt[buf], 0, 72, 32);  s0 = WMMA_F16(qa1, b1, s0);
      v16h b2 = ld_frag(Kt[buf], 16, 72, 0);  s1 = WMMA_F16(qa0, b2, s1);
      v16h b3 = ld_frag(Kt[buf], 16, 72, 32); s1 = WMMA_F16(qa1, b3, s1);
    }

    _Float16* pw = Pw[wave];
#pragma unroll
    for (int r = 0; r < 8; ++r) {
      float a0 = s0[r] * 0.125f;   // 1/sqrt(64)
      float a1 = s1[r] * 0.125f;
      float mx = fmaxf(a0, a1);
      mx = fmaxf(mx, __shfl_xor(mx, 1, 32));
      mx = fmaxf(mx, __shfl_xor(mx, 2, 32));
      mx = fmaxf(mx, __shfl_xor(mx, 4, 32));
      mx = fmaxf(mx, __shfl_xor(mx, 8, 32));
      const float mnew = fmaxf(mrow[r], mx);
      const float corr = __expf(mrow[r] - mnew);
      mrow[r] = mnew;
      const float p0 = __expf(a0 - mnew);
      const float p1 = __expf(a1 - mnew);
      float rs = p0 + p1;
      rs += __shfl_xor(rs, 1, 32);
      rs += __shfl_xor(rs, 2, 32);
      rs += __shfl_xor(rs, 4, 32);
      rs += __shfl_xor(rs, 8, 32);
      lsum[r] = lsum[r] * corr + rs;
      o0[r] *= corr; o1[r] *= corr; o2[r] *= corr; o3[r] *= corr;
      const int prow = (r + 8 * lh) * 40;   // C-layout row -> LDS row (A layout)
      pw[prow + nl]      = (_Float16)p0;
      pw[prow + 16 + nl] = (_Float16)p1;
    }
    asm volatile("s_wait_dscnt 0" ::: "memory");  // in-wave P store->frag-load fence

    const v16h pa = ld_frag(pw, 0, 40, 0);
    v16h vb0 = ld_frag(Vt[buf], 0, 48, 0);   o0 = WMMA_F16(pa, vb0, o0);
    v16h vb1 = ld_frag(Vt[buf], 16, 48, 0);  o1 = WMMA_F16(pa, vb1, o1);
    v16h vb2 = ld_frag(Vt[buf], 32, 48, 0);  o2 = WMMA_F16(pa, vb2, o2);
    v16h vb3 = ld_frag(Vt[buf], 48, 48, 0);  o3 = WMMA_F16(pa, vb3, o3);

    wait_async0();       // next tile's LDS writes complete (this wave)
    __syncthreads();     // all waves done reading buf & writing buf^1
  }

#pragma unroll
  for (int r = 0; r < 8; ++r) {
    const float inv = 1.0f / lsum[r];
    const int s = qt * 64 + wave * 16 + r + 8 * lh;
    _Float16* orow = Oh + ((size_t)b * SEQLEN + s) * DMODEL + h * DK;
    orow[0 * 16 + nl] = (_Float16)(o0[r] * inv);
    orow[1 * 16 + nl] = (_Float16)(o1[r] * inv);
    orow[2 * 16 + nl] = (_Float16)(o2[r] * inv);
    orow[3 * 16 + nl] = (_Float16)(o3[r] * inv);
  }
}

// ---------------------------------------------------------------------------
// Kernel 3: out = O @ Wout^T + bout   (M=4096, N=1024, K=1024), fp32 out.
// A tile (already f16) filled with async global->LDS copies that overlap the
// f32->f16 converting fill of the B tile.
// ---------------------------------------------------------------------------
__global__ __launch_bounds__(128) void out_gemm_kernel(
    const _Float16* __restrict__ Oh, const float* __restrict__ Wout,
    const float* __restrict__ bout, float* __restrict__ out) {
  __shared__ _Float16 At[64 * 48];   // pitch 48 halfs = 96B (16B aligned rows)
  __shared__ _Float16 Bt[64 * 40];
  const int tid = threadIdx.x, wave = tid >> 5, lane = tid & 31;
  const int m0 = blockIdx.x * 64, n0 = blockIdx.y * 64;
  const int lrow = tid >> 1, lseg = tid & 1;
  v8f acc[4] = {};

  for (int k0 = 0; k0 < DMODEL; k0 += 32) {
    __syncthreads();
    async_cp16h(&At[lrow * 48 + lseg * 16],
                Oh + (size_t)(m0 + lrow) * DMODEL + k0 + lseg * 16);
    cvt_store16(&Bt[lrow * 40 + lseg * 16],
                (const float4*)(Wout + (size_t)(n0 + lrow) * DMODEL + k0 + lseg * 16));
    wait_async0();
    __syncthreads();
    v16h a = ld_frag(At, wave * 16, 48, 0);
#pragma unroll
    for (int j = 0; j < 4; ++j) {
      v16h b = ld_frag(Bt, j * 16, 40, 0);
      acc[j] = WMMA_F16(a, b, acc[j]);
    }
  }

  const int lh = lane >> 4, nl = lane & 15;
#pragma unroll
  for (int j = 0; j < 4; ++j) {
    const int n = n0 + j * 16 + nl;
    const float bias = bout[n];
#pragma unroll
    for (int r = 0; r < 8; ++r) {
      const int m = m0 + wave * 16 + r + 8 * lh;
      out[(size_t)m * DMODEL + n] = acc[j][r] + bias;
    }
  }
}

// ---------------------------------------------------------------------------
extern "C" void kernel_launch(void* const* d_in, const int* in_sizes, int n_in,
                              void* d_out, int out_size, void* d_ws, size_t ws_size,
                              hipStream_t stream) {
  const float* x    = (const float*)d_in[0];
  const float* Wqkv = (const float*)d_in[1];
  const float* bqkv = (const float*)d_in[2];
  const float* Wout = (const float*)d_in[3];
  const float* bout = (const float*)d_in[4];
  float* out = (float*)d_out;

  char* ws = (char*)d_ws;
  const size_t headBuf = (size_t)NBATCH * NHEADS * SEQLEN * DK * sizeof(_Float16); // 8 MiB
  _Float16* Qh  = (_Float16*)(ws);
  _Float16* Kh  = (_Float16*)(ws + headBuf);
  _Float16* Vth = (_Float16*)(ws + 2 * headBuf);
  _Float16* Oh  = (_Float16*)(ws + 3 * headBuf);

  qkv_gemm_kernel<<<dim3(64, 48), 128, 0, stream>>>(x, Wqkv, bqkv, Qh, Kh, Vth);
  attn_kernel<<<dim3(NBATCH * NHEADS * (SEQLEN / 64)), 128, 0, stream>>>(Qh, Kh, Vth, Oh);
  out_gemm_kernel<<<dim3(64, 16), 128, 0, stream>>>(Oh, Wout, bout, out);
}
